// EdgeGCN_566935683373
// MI455X (gfx1250) — matrix operand order repaired and checked by reading, not compiled, not run on hardware
//
#include <hip/hip_runtime.h>
#include <math.h>

// ---- CDNA5 WMMA vector types (wave32) ----
typedef __attribute__((ext_vector_type(16))) _Float16 v16h;
typedef __attribute__((ext_vector_type(8)))  float    v8f;

#define N_NODES  100000
#define N_EDGES  3200000
#define N_FEAT   128
#define HID      16
#define N_DECODE 500000
#define COS_EPS  1e-6f

#define N_TILES  (N_NODES / HID)   // 6250, exact
#define TPW      4                 // row-tiles per wave (B-register reuse)
#define NW_GEMM  ((N_TILES + TPW - 1) / TPW)   // 1563 waves

// ---------------- degree / norm ----------------
__global__ void init_deg_kernel(float* __restrict__ deg) {
    int i = blockIdx.x * blockDim.x + threadIdx.x;
    if (i < N_NODES) deg[i] = 1.0f;            // self-loop contributes 1
}

__global__ void accum_deg_kernel(float* __restrict__ deg,
                                 const long long* __restrict__ col) {
    int e = blockIdx.x * blockDim.x + threadIdx.x;
    if (e < N_EDGES) atomicAdd(&deg[(int)col[e]], 1.0f);
}

__global__ void dinv_kernel(float* __restrict__ deg) {
    int i = blockIdx.x * blockDim.x + threadIdx.x;
    if (i < N_NODES) deg[i] = rsqrtf(deg[i]);  // deg >= 1 always
}

// ---------------- GEMM1: h1pre[N,16] = x[N,128] @ W1[128,16] (WMMA f16->f32) ----
// W1 staged once per block into LDS as f16; each wave computes TPW 16-row tiles,
// reusing the 4 B-fragments (32 VGPRs) across all tiles.
__global__ __launch_bounds__(256)
void gemm1_wmma_kernel(const float* __restrict__ x,
                       const float* __restrict__ W1,
                       float* __restrict__ h1pre) {
    __shared__ _Float16 sW[N_FEAT * HID];      // 4 KB of the 320 KB WGP LDS
    for (int i = threadIdx.x; i < N_FEAT * HID; i += blockDim.x)
        sW[i] = (_Float16)W1[i];
    __syncthreads();

    int gw   = (blockIdx.x * blockDim.x + threadIdx.x) >> 5;   // global wave id
    int lane = threadIdx.x & 31;
    if (gw >= NW_GEMM) return;                 // whole-wave exit (after barrier)

    int nn = lane & 15;
    int hi = lane >> 4;

    // Build all four 16x32 B-fragments from LDS (K = kb + 16*hi + e)
    v16h bmat[4];
    #pragma unroll
    for (int kb4 = 0; kb4 < 4; ++kb4) {
        #pragma unroll
        for (int e = 0; e < 16; ++e)
            bmat[kb4][e] = sW[(kb4 * 32 + 16 * hi + e) * HID + nn];
    }

    int t0   = gw * TPW;
    int tend = (t0 + TPW < N_TILES) ? (t0 + TPW) : N_TILES;    // wave-uniform
    for (int t = t0; t < tend; ++t) {
        int m0 = t * 16;
        int m  = m0 + nn;                                      // A-row for this lane
        if (t + 1 < N_TILES)                                   // stream-ahead hint
            __builtin_prefetch(&x[(size_t)(m + 16) * N_FEAT], 0, 1);
        v8f c = {};
        #pragma unroll
        for (int kb4 = 0; kb4 < 4; ++kb4) {
            int kb = kb4 * 32;
            v16h a;
            // A 16x32 f16, contiguous k order (same as B): pairs p -> K = 16*hi + 2p
            #pragma unroll
            for (int p = 0; p < 8; ++p) {
                int ka = 16 * hi + 2 * p;
                a[2 * p]     = (_Float16)x[(size_t)m * N_FEAT + kb + ka];
                a[2 * p + 1] = (_Float16)x[(size_t)m * N_FEAT + kb + ka + 1];
            }
            c = __builtin_amdgcn_wmma_f32_16x16x32_f16(false, a, false, bmat[kb4],
                                                       (short)0, c, false, false);
        }
        // C/D: lane half hi, VGPR v -> row m0 + v + 8*hi, col nn
        #pragma unroll
        for (int v = 0; v < 8; ++v)
            h1pre[(size_t)(m0 + v + 8 * hi) * HID + nn] = c[v];
    }
}

// ---------------- GEMM2: h2pre[N,16] = h1[N,16] @ W2[16,16], K padded 16->32 ----
__global__ __launch_bounds__(256)
void gemm2_wmma_kernel(const float* __restrict__ h1,
                       const float* __restrict__ W2,
                       float* __restrict__ h2pre) {
    __shared__ _Float16 sW[HID * HID];         // 512 B
    if (threadIdx.x < HID * HID)
        sW[threadIdx.x] = (_Float16)W2[threadIdx.x];
    __syncthreads();

    int gw   = (blockIdx.x * blockDim.x + threadIdx.x) >> 5;
    int lane = threadIdx.x & 31;
    if (gw >= NW_GEMM) return;

    int nn = lane & 15;
    int hi = lane >> 4;

    v16h b;
    #pragma unroll
    for (int e = 0; e < 16; ++e) {             // K = (e<8 ? 8*hi+e : 16+8*hi+e-8); real only for K<16
        int kbb = (e < 8) ? (8 * hi + e) : (16 + 8 * hi + (e - 8));
        b[e] = (kbb < HID) ? sW[kbb * HID + nn] : (_Float16)0.0f;
    }

    int t0   = gw * TPW;
    int tend = (t0 + TPW < N_TILES) ? (t0 + TPW) : N_TILES;
    for (int t = t0; t < tend; ++t) {
        int m0 = t * 16;
        int m  = m0 + nn;
        v16h a = {};                           // K >= 16 stays zero
        #pragma unroll
        for (int p = 0; p < 4; ++p) {          // K = 8*hi + 2p in [0,16)
            int ka = 8 * hi + 2 * p;
            a[2 * p]     = (_Float16)h1[(size_t)m * HID + ka];
            a[2 * p + 1] = (_Float16)h1[(size_t)m * HID + ka + 1];
        }
        v8f c = {};
        c = __builtin_amdgcn_wmma_f32_16x16x32_f16(false, a, false, b,
                                                   (short)0, c, false, false);
        #pragma unroll
        for (int v = 0; v < 8; ++v)
            h2pre[(size_t)(m0 + v + 8 * hi) * HID + nn] = c[v];
    }
}

// ---------------- aggregation ----------------
// agg[i,j] = hpre[i,j] * dinv[i]^2   (the self-loop message, no atomics needed)
__global__ void init_agg_kernel(float* __restrict__ agg,
                                const float* __restrict__ hpre,
                                const float* __restrict__ dinv) {
    int i = blockIdx.x * blockDim.x + threadIdx.x;
    if (i < N_NODES * HID) {
        float dv = dinv[i >> 4];
        agg[i] = hpre[i] * dv * dv;
    }
}

// One edge per lane: coalesced int64 index loads, b128 message-row loads,
// 16 f32 atomics per edge (51.2M total per layer -- the hard floor).
__global__ void scatter_kernel(float* __restrict__ agg,
                               const float* __restrict__ hpre,
                               const float* __restrict__ dinv,
                               const long long* __restrict__ row,
                               const long long* __restrict__ col) {
    int e = blockIdx.x * blockDim.x + threadIdx.x;
    if (e >= N_EDGES) return;
    int r = (int)row[e];
    int c = (int)col[e];
    float nrm = dinv[r] * dinv[c];
    const float4* src = (const float4*)(hpre + (size_t)r * HID);
    float*        dst = agg + (size_t)c * HID;
    #pragma unroll
    for (int q = 0; q < 4; ++q) {
        float4 v = src[q];
        atomicAdd(dst + 4 * q + 0, v.x * nrm);
        atomicAdd(dst + 4 * q + 1, v.y * nrm);
        atomicAdd(dst + 4 * q + 2, v.z * nrm);
        atomicAdd(dst + 4 * q + 3, v.w * nrm);
    }
}

__global__ void finalize_relu_kernel(float* __restrict__ agg,
                                     const float* __restrict__ bias) {
    int i = blockIdx.x * blockDim.x + threadIdx.x;
    if (i < N_NODES * HID) agg[i] = fmaxf(agg[i] + bias[i & 15], 0.0f);
}

__global__ void finalize_bias_kernel(float* __restrict__ agg,
                                     const float* __restrict__ bias) {
    int i = blockIdx.x * blockDim.x + threadIdx.x;
    if (i < N_NODES * HID) agg[i] = agg[i] + bias[i & 15];
}

// ---------------- decode: sigmoid(cos_sim(h2[a], h2[c])) ----------------
__global__ void decode_kernel(const float* __restrict__ h2,
                              const long long* __restrict__ didx,
                              float* __restrict__ out) {
    int d = blockIdx.x * blockDim.x + threadIdx.x;
    if (d >= N_DECODE) return;
    int ia = (int)didx[d];
    int ic = (int)didx[N_DECODE + d];
    const float4* pa = (const float4*)(h2 + (size_t)ia * HID);
    const float4* pc = (const float4*)(h2 + (size_t)ic * HID);
    float num = 0.0f, na = 0.0f, nc = 0.0f;
    #pragma unroll
    for (int q = 0; q < 4; ++q) {
        float4 av = pa[q];
        float4 cv = pc[q];
        num += av.x * cv.x + av.y * cv.y + av.z * cv.z + av.w * cv.w;
        na  += av.x * av.x + av.y * av.y + av.z * av.z + av.w * av.w;
        nc  += cv.x * cv.x + cv.y * cv.y + cv.z * cv.z + cv.w * cv.w;
    }
    float den = fmaxf(sqrtf(na) * sqrtf(nc), COS_EPS);
    float z = num / den;
    out[d] = 1.0f / (1.0f + expf(-z));
}

// ---------------- launch ----------------
extern "C" void kernel_launch(void* const* d_in, const int* in_sizes, int n_in,
                              void* d_out, int out_size, void* d_ws, size_t ws_size,
                              hipStream_t stream) {
    (void)in_sizes; (void)n_in; (void)out_size; (void)ws_size;
    const float*     x   = (const float*)d_in[0];
    const long long* ei  = (const long long*)d_in[1];  // [2, E] int64 per reference
    const long long* di  = (const long long*)d_in[2];  // [2, D] int64
    const float*     W1  = (const float*)d_in[3];
    const float*     b1  = (const float*)d_in[4];
    const float*     W2  = (const float*)d_in[5];
    const float*     b2  = (const float*)d_in[6];
    float*           out = (float*)d_out;

    const long long* row = ei;             // edge_index[0] (source)
    const long long* col = ei + N_EDGES;   // edge_index[1] (target)

    // workspace layout (256B-aligned blocks)
    char* base = (char*)d_ws;
    size_t off = 0;
    auto carve = [&](size_t bytes) -> float* {
        float* p = (float*)(base + off);
        off += (bytes + 255) & ~(size_t)255;
        return p;
    };
    float* dinv  = carve((size_t)N_NODES * sizeof(float));        // deg -> dinv in place
    float* h1pre = carve((size_t)N_NODES * HID * sizeof(float));
    float* agg1  = carve((size_t)N_NODES * HID * sizeof(float));  // becomes h1 in place
    float* h2pre = carve((size_t)N_NODES * HID * sizeof(float));
    float* agg2  = carve((size_t)N_NODES * HID * sizeof(float));  // becomes h2 in place

    const int TPB = 256;
    int gN  = (N_NODES + TPB - 1) / TPB;
    int gE  = (N_EDGES + TPB - 1) / TPB;
    int gNH = (N_NODES * HID + TPB - 1) / TPB;
    int gG  = (NW_GEMM * 32 + TPB - 1) / TPB;
    int gD  = (N_DECODE + TPB - 1) / TPB;

    // normalization
    init_deg_kernel<<<gN, TPB, 0, stream>>>(dinv);
    accum_deg_kernel<<<gE, TPB, 0, stream>>>(dinv, col);
    dinv_kernel<<<gN, TPB, 0, stream>>>(dinv);

    // layer 1
    gemm1_wmma_kernel<<<gG, TPB, 0, stream>>>(x, W1, h1pre);
    init_agg_kernel<<<gNH, TPB, 0, stream>>>(agg1, h1pre, dinv);
    scatter_kernel<<<gE, TPB, 0, stream>>>(agg1, h1pre, dinv, row, col);
    finalize_relu_kernel<<<gNH, TPB, 0, stream>>>(agg1, b1);

    // layer 2
    gemm2_wmma_kernel<<<gG, TPB, 0, stream>>>(agg1, W2, h2pre);
    init_agg_kernel<<<gNH, TPB, 0, stream>>>(agg2, h2pre, dinv);
    scatter_kernel<<<gE, TPB, 0, stream>>>(agg2, h2pre, dinv, row, col);
    finalize_bias_kernel<<<gNH, TPB, 0, stream>>>(agg2, b2);

    // decode
    decode_kernel<<<gD, TPB, 0, stream>>>(agg2, di, out);
}
